// UnitSpatialGcn_43439299231763
// MI455X (gfx1250) — matrix operand order repaired and checked by reading, not compiled, not run on hardware
//
#include <hip/hip_runtime.h>
#include <hip/hip_bf16.h>

// ---------------------------------------------------------------------------
// UnitSpatialGcn for MI455X (gfx1250), bf16 WMMA (v_wmma_f32_16x16x32_bf16).
// N=8, C=64, COUT=128, T=V=256, S=3, IC=32.
// Channels-last layouts so every WMMA fragment is contiguous in K:
//   A frag = 2x 16B loads + shuffle, B frag = 1x 32B vector load.
// ---------------------------------------------------------------------------

typedef __attribute__((ext_vector_type(16))) __bf16 v16bf;
typedef __attribute__((ext_vector_type(8)))  __bf16 v8bf;
typedef __attribute__((ext_vector_type(8)))  float  v8f;

#define N_  8
#define C_  64
#define CO_ 128
#define T_  256
#define V_  256
#define IC_ 32
#define TV_ 65536

__device__ __forceinline__ v8f wmma_bf(v16bf a, v16bf b, v8f c) {
  return __builtin_amdgcn_wmma_f32_16x16x32_bf16(false, a, false, b,
                                                 (short)0, c, false, false);
}

// A-fragment (16x32 bf16 MxK): lane holds row m=lane&15; half=lane>>4.
// K for element e: 16*half+e
// => two contiguous 8-element chunks.
__device__ __forceinline__ v16bf load_a16(const __bf16* p, int half) {
  v8bf lo = *(const v8bf*)(p + (half << 4));
  v8bf hi = *(const v8bf*)(p + 8 + (half << 4));
  return __builtin_shufflevector(lo, hi, 0, 1, 2, 3, 4, 5, 6, 7,
                                 8, 9, 10, 11, 12, 13, 14, 15);
}
// B-fragment (32x16 bf16 KxN): lane holds col n=lane&15; K = 16*half + e
// => one contiguous 16-element chunk.
__device__ __forceinline__ v16bf load_b16(const __bf16* p, int half) {
  return *(const v16bf*)(p + (half << 4));
}

// ---------------------------------------------------------------------------
// 1) x [N,C,T,V] fp32 -> xc bf16 [N,TV,C] (channels-last) and xbt [N,C,V,T]
// ---------------------------------------------------------------------------
__global__ __launch_bounds__(256) void k_cvt(const float* __restrict__ x,
                                             __bf16* __restrict__ xc,
                                             __bf16* __restrict__ xbt) {
  long i = (long)blockIdx.x * 256 + threadIdx.x;
  long total = (long)N_ * C_ * TV_;
  for (; i < total; i += (long)gridDim.x * 256) {
    float v = x[i];
    int tv = (int)(i & (TV_ - 1));
    long nc = i >> 16;               // n*64 + c
    int n = (int)(nc >> 6), c = (int)(nc & 63);
    int t = tv >> 8, vv = tv & 255;
    xc[((long)n * TV_ + tv) * C_ + c] = (__bf16)v;
    xbt[nc * TV_ + (long)vv * 256 + t] = (__bf16)v;
  }
}

// generic fp32 -> bf16 weight convert (row-major preserved, K contiguous)
__global__ __launch_bounds__(256) void k_w2bf(const float* __restrict__ s,
                                              __bf16* __restrict__ d, int n) {
  int i = blockIdx.x * 256 + threadIdx.x;
  if (i < n) d[i] = (__bf16)s[i];
}

// tcn_w [o][c][kh] -> bf16 [o][kh][c]  (K-order kh*128+c)
__global__ __launch_bounds__(256) void k_twcvt(const float* __restrict__ s,
                                               __bf16* __restrict__ d) {
  int i = blockIdx.x * 256 + threadIdx.x;   // 147456 total
  int o = i / 1152, r = i % 1152;
  int kh = r >> 7, c = r & 127;
  d[i] = (__bf16)s[(o * CO_ + c) * 9 + kh];
}

// ---------------------------------------------------------------------------
// 2) embed: out[n,t,v,m] = sum_c W[m,c]*x[n,c,t,v] + bias[m]  (M=32, K=64)
//    output [N,T,V,IC] => scores K-order is (v,c), same on both operands.
// ---------------------------------------------------------------------------
__global__ __launch_bounds__(256) void k_embed(const __bf16* __restrict__ W,
                                               const float* __restrict__ bias,
                                               const __bf16* __restrict__ xc,
                                               __bf16* __restrict__ out) {
  int wv = threadIdx.x >> 5, lane = threadIdx.x & 31;
  int half = lane >> 4, l16 = lane & 15;
  int n = blockIdx.x >> 10;
  int j0 = ((blockIdx.x & 1023) << 6) + ((wv >> 1) << 4);
  int m_tile = wv & 1;
  const __bf16* arow = W + ((m_tile << 4) + l16) * C_;
  const __bf16* bptr = xc + ((long)n * TV_ + j0 + l16) * C_;
  v8f acc = {};
#pragma unroll
  for (int kk = 0; kk < C_; kk += 32)
    acc = wmma_bf(load_a16(arow + kk, half), load_b16(bptr + kk, half), acc);
  int j = j0 + l16;
  int m0 = (m_tile << 4) + (half << 3);
  v8f bv = *(const v8f*)(bias + m0);
  v8bf ov;
#pragma unroll
  for (int r = 0; r < 8; ++r) ov[r] = (__bf16)(acc[r] + bv[r]);
  *(v8bf*)(out + ((long)n * TV_ + j) * IC_ + m0) = ov;
}

// ---------------------------------------------------------------------------
// 3) scores: sc[n,s,t] = (1/V) * sum_k ap[n,t,k]*bp[n,s,k],  K = V*IC = 8192
// ---------------------------------------------------------------------------
__global__ __launch_bounds__(256) void k_scores(const __bf16* __restrict__ ap,
                                                const __bf16* __restrict__ bp,
                                                float* __restrict__ sc) {
  int wv = threadIdx.x >> 5, lane = threadIdx.x & 31;
  int half = lane >> 4, l16 = lane & 15;
  int n = blockIdx.x >> 5, rem = blockIdx.x & 31;
  int t0 = (rem >> 3) * 64 + (wv >> 1) * 16;
  int s0 = (rem & 7) * 32 + (wv & 1) * 16;
  const __bf16* abase = ap + ((long)n * T_ + t0 + l16) * 8192;
  const __bf16* bbase = bp + ((long)n * T_ + s0 + l16) * 8192;
  v8f acc = {};
  for (int kk = 0; kk < 8192; kk += 32) {
    __builtin_prefetch(abase + kk + 512, 0, 0);
    __builtin_prefetch(bbase + kk + 512, 0, 0);
    acc = wmma_bf(load_a16(abase + kk, half), load_b16(bbase + kk, half), acc);
  }
  int s = s0 + l16;
  v8f ov = acc * (1.0f / 256.0f);
  *(v8f*)(sc + ((long)n * T_ + s) * T_ + t0 + (half << 3)) = ov;
}

// ---------------------------------------------------------------------------
// 4) adT[n,s,t] = bf16( A_static[t,s] + softmax_t(scores)[n,t,s] )
//    block per (n,s); coalesced row reads; LDS tree reductions over t.
// ---------------------------------------------------------------------------
__global__ __launch_bounds__(256) void k_softmax(const float* __restrict__ sc,
                                                 const float* __restrict__ Am,
                                                 const float* __restrict__ Aw,
                                                 __bf16* __restrict__ adT) {
  __shared__ float red[256];
  int t = threadIdx.x;
  int n = blockIdx.x >> 8, s = blockIdx.x & 255;
  float val = sc[((long)n * T_ + s) * T_ + t];
  red[t] = val; __syncthreads();
  for (int o = 128; o > 0; o >>= 1) {
    if (t < o) red[t] = fmaxf(red[t], red[t + o]);
    __syncthreads();
  }
  float mx = red[0]; __syncthreads();
  float e = __expf(val - mx);
  red[t] = e; __syncthreads();
  for (int o = 128; o > 0; o >>= 1) {
    if (t < o) red[t] += red[t + o];
    __syncthreads();
  }
  float att = e / red[0];
  adT[((long)n * T_ + s) * T_ + t] =
      (__bf16)(Am[t * T_ + s] + Aw[t * T_ + s] + att);
}

// ---------------------------------------------------------------------------
// 5) xa[n,(v,s),c] = sum_t xbt[n,c,v,t] * adapt[n,t,s]  (per (n,c) 256^3 GEMM)
// ---------------------------------------------------------------------------
__global__ __launch_bounds__(256) void k_xa(const __bf16* __restrict__ xbt,
                                            const __bf16* __restrict__ adT,
                                            __bf16* __restrict__ xa) {
  int wv = threadIdx.x >> 5, lane = threadIdx.x & 31;
  int half = lane >> 4, l16 = lane & 15;
  int nc = blockIdx.x >> 5, rem = blockIdx.x & 31;
  int v0 = (rem >> 3) * 64 + (wv >> 1) * 16;
  int s0 = (rem & 7) * 32 + (wv & 1) * 16;
  int n = nc >> 6, c = nc & 63;
  const __bf16* abase = xbt + ((long)nc * 256 + v0 + l16) * 256;
  const __bf16* bbase = adT + ((long)n * T_ + s0 + l16) * 256;
  v8f acc = {};
#pragma unroll 2
  for (int kk = 0; kk < 256; kk += 32)
    acc = wmma_bf(load_a16(abase + kk, half), load_b16(bbase + kk, half), acc);
  int s = s0 + l16;
#pragma unroll
  for (int r = 0; r < 8; ++r) {
    int v = v0 + (half << 3) + r;
    xa[((long)n * TV_ + v * 256 + s) * C_ + c] = (__bf16)acc[r];
  }
}

// ---------------------------------------------------------------------------
// 6) y[n,j,o] (+)= sum_c wd[o,c]*xa[n,j,c] + bd[o]   (M=128, K=64, CL fp32)
// ---------------------------------------------------------------------------
__global__ __launch_bounds__(256) void k_y(const __bf16* __restrict__ wd,
                                           const float* __restrict__ bd,
                                           const __bf16* __restrict__ xa,
                                           float* __restrict__ y, int accum) {
  int wv = threadIdx.x >> 5, lane = threadIdx.x & 31;
  int half = lane >> 4, l16 = lane & 15;
  int n = blockIdx.x >> 12;
  int j0 = (blockIdx.x & 4095) << 4;
  const __bf16* arow = wd + ((wv << 4) + l16) * C_;
  const __bf16* bptr = xa + ((long)n * TV_ + j0 + l16) * C_;
  v8f acc = {};
#pragma unroll
  for (int kk = 0; kk < C_; kk += 32)
    acc = wmma_bf(load_a16(arow + kk, half), load_b16(bptr + kk, half), acc);
  int m0 = (wv << 4) + (half << 3);
  float* base = y + ((long)n * TV_ + j0 + l16) * CO_ + m0;
  v8f bv = *(const v8f*)(bd + m0);
  v8f ov = acc + bv;
  if (accum) ov += *(const v8f*)base;
  *(v8f*)base = ov;
}

// ---------------------------------------------------------------------------
// 7) gcn[n,j,o] = bf16(relu( BN_g(y) + BN_c(cres_conv(x)) ))  fused GEMM
// ---------------------------------------------------------------------------
__global__ __launch_bounds__(256) void k_gcn(const __bf16* __restrict__ cw,
                                             const float* __restrict__ cb,
                                             const float* __restrict__ bng,
                                             const float* __restrict__ bnc,
                                             const __bf16* __restrict__ xc,
                                             const float* __restrict__ y,
                                             __bf16* __restrict__ gcn) {
  int wv = threadIdx.x >> 5, lane = threadIdx.x & 31;
  int half = lane >> 4, l16 = lane & 15;
  int n = blockIdx.x >> 12;
  int j0 = (blockIdx.x & 4095) << 4;
  const __bf16* arow = cw + ((wv << 4) + l16) * C_;
  const __bf16* bptr = xc + ((long)n * TV_ + j0 + l16) * C_;
  v8f acc = {};
#pragma unroll
  for (int kk = 0; kk < C_; kk += 32)
    acc = wmma_bf(load_a16(arow + kk, half), load_b16(bptr + kk, half), acc);
  int m0 = (wv << 4) + (half << 3);
  long base = ((long)n * TV_ + j0 + l16) * CO_ + m0;
  v8f yv = *(const v8f*)(y + base);
  v8f gG = *(const v8f*)(bng + m0), gB = *(const v8f*)(bng + 128 + m0);
  v8f gM = *(const v8f*)(bng + 256 + m0), gV = *(const v8f*)(bng + 384 + m0);
  v8f cG = *(const v8f*)(bnc + m0), cB = *(const v8f*)(bnc + 128 + m0);
  v8f cM = *(const v8f*)(bnc + 256 + m0), cV = *(const v8f*)(bnc + 384 + m0);
  v8f cb8 = *(const v8f*)(cb + m0);
  v8bf ov;
#pragma unroll
  for (int r = 0; r < 8; ++r) {
    float g1 = (yv[r] - gM[r]) * (gG[r] * rsqrtf(gV[r] + 1e-5f)) + gB[r];
    float g2 = (acc[r] + cb8[r] - cM[r]) * (cG[r] * rsqrtf(cV[r] + 1e-5f)) + cB[r];
    ov[r] = (__bf16)fmaxf(g1 + g2, 0.0f);
  }
  *(v8bf*)(gcn + base) = ov;
}

// ---------------------------------------------------------------------------
// 8) temporal (9,1) conv pad 4, implicit im2col: K = 9*128, k = kh*128 + c
//    => kh (and padded row hh) uniform per 32-wide k-step: one vector B load.
// ---------------------------------------------------------------------------
__global__ __launch_bounds__(256) void k_tcn(const __bf16* __restrict__ tw,
                                             const __bf16* __restrict__ gcn,
                                             float* __restrict__ tpre) {
  int wv = threadIdx.x >> 5, lane = threadIdx.x & 31;
  int half = lane >> 4, l16 = lane & 15;
  int n = blockIdx.x >> 12;
  int j0 = (blockIdx.x & 4095) << 4;
  int j = j0 + l16, h = j >> 8, w = j & 255;
  const __bf16* arow = tw + ((wv << 4) + l16) * 1152;
  const __bf16* gb = gcn + (long)n * TV_ * CO_;
  v8f acc = {};
  for (int kk = 0; kk < 1152; kk += 32) {
    int kh = kk >> 7, c0 = kk & 127;
    int hh = h + kh - 4;
    __builtin_prefetch(arow + kk + 64, 0, 0);
    v16bf bf_ = {};
    if ((unsigned)hh < 256u)
      bf_ = *(const v16bf*)(gb + ((long)hh * 256 + w) * CO_ + c0 + (half << 4));
    acc = wmma_bf(load_a16(arow + kk, half), bf_, acc);
  }
  *(v8f*)(tpre + ((long)n * TV_ + j) * CO_ + (wv << 4) + (half << 3)) = acc;
}

// ---------------------------------------------------------------------------
// 9) out[n,o,j] = relu( BN_t(tpre + tcn_b) + BN_r(rt_conv(x) + rt_b) )
// ---------------------------------------------------------------------------
__global__ __launch_bounds__(256) void k_final(const __bf16* __restrict__ rw,
                                               const float* __restrict__ rb,
                                               const float* __restrict__ bnt,
                                               const float* __restrict__ tb,
                                               const float* __restrict__ bnr,
                                               const __bf16* __restrict__ xc,
                                               const float* __restrict__ tpre,
                                               float* __restrict__ out) {
  int wv = threadIdx.x >> 5, lane = threadIdx.x & 31;
  int half = lane >> 4, l16 = lane & 15;
  int n = blockIdx.x >> 12;
  int j0 = (blockIdx.x & 4095) << 4;
  const __bf16* arow = rw + ((wv << 4) + l16) * C_;
  const __bf16* bptr = xc + ((long)n * TV_ + j0 + l16) * C_;
  v8f acc = {};
#pragma unroll
  for (int kk = 0; kk < C_; kk += 32)
    acc = wmma_bf(load_a16(arow + kk, half), load_b16(bptr + kk, half), acc);
  int j = j0 + l16;
  int m0 = (wv << 4) + (half << 3);
  v8f tp = *(const v8f*)(tpre + ((long)n * TV_ + j) * CO_ + m0);
  v8f tG = *(const v8f*)(bnt + m0), tB = *(const v8f*)(bnt + 128 + m0);
  v8f tM = *(const v8f*)(bnt + 256 + m0), tVv = *(const v8f*)(bnt + 384 + m0);
  v8f rG = *(const v8f*)(bnr + m0), rB = *(const v8f*)(bnr + 128 + m0);
  v8f rM = *(const v8f*)(bnr + 256 + m0), rV = *(const v8f*)(bnr + 384 + m0);
  v8f tb8 = *(const v8f*)(tb + m0), rb8 = *(const v8f*)(rb + m0);
#pragma unroll
  for (int r = 0; r < 8; ++r) {
    float tv = (tp[r] + tb8[r] - tM[r]) * (tG[r] * rsqrtf(tVv[r] + 1e-5f)) + tB[r];
    float rv = (acc[r] + rb8[r] - rM[r]) * (rG[r] * rsqrtf(rV[r] + 1e-5f)) + rB[r];
    out[((long)n * CO_ + m0 + r) * TV_ + j] = fmaxf(tv + rv, 0.0f);
  }
}

// ---------------------------------------------------------------------------
extern "C" void kernel_launch(void* const* d_in, const int* in_sizes, int n_in,
                              void* d_out, int out_size, void* d_ws, size_t ws_size,
                              hipStream_t stream) {
  const float* x       = (const float*)d_in[0];
  const float* mat_adj = (const float*)d_in[1];
  const float* adj_w   = (const float*)d_in[2];
  const float* wa      = (const float*)d_in[3];
  const float* ba      = (const float*)d_in[4];
  const float* wb      = (const float*)d_in[5];
  const float* bb      = (const float*)d_in[6];
  const float* wd      = (const float*)d_in[7];
  const float* bd      = (const float*)d_in[8];
  const float* gcn_bn  = (const float*)d_in[9];
  const float* cres_w  = (const float*)d_in[10];
  const float* cres_b  = (const float*)d_in[11];
  const float* cres_bn = (const float*)d_in[12];
  const float* tcn_w   = (const float*)d_in[13];
  const float* tcn_b   = (const float*)d_in[14];
  const float* tcn_bn  = (const float*)d_in[15];
  const float* rt_w    = (const float*)d_in[16];
  const float* rt_b    = (const float*)d_in[17];
  const float* rt_bn   = (const float*)d_in[18];
  float* out = (float*)d_out;

  char* ws = (char*)d_ws;
  __bf16* xc   = (__bf16*)(ws + 0);           //  64 MB  x bf16 [N,TV,C]
  __bf16* xbt  = (__bf16*)(ws + 67108864);    //  64 MB  x bf16 [N,C,V,T]
  __bf16* ap   = (__bf16*)(ws + 134217728);   //  32 MB  embed a [N,T,V,IC]
  __bf16* bp   = (__bf16*)(ws + 167772160);   //  32 MB  embed b [N,T,V,IC]
  float*  sc   = (float*) (ws + 201326592);   //   2 MB  scores [N,s,t]
  __bf16* adT  = (__bf16*)(ws + 203423744);   //   1 MB  adapt  [N,s,t]
  __bf16* xa   = (__bf16*)(ws + 204472320);   //  64 MB  xa     [N,TV,C]
  float*  y    = (float*) (ws + 271581184);   // 256 MB  y fp32 [N,TV,CO]
  __bf16* gcn  = (__bf16*)(ws + 540016640);   // 128 MB  gcn    [N,TV,CO]
  __bf16* wab  = (__bf16*)(ws + 674234368);   // bf16 weights below
  __bf16* wbb  = wab + 3 * IC_ * C_;
  __bf16* wdb  = wbb + 3 * IC_ * C_;
  __bf16* cwb  = wdb + 3 * CO_ * C_;
  __bf16* rwb  = cwb + CO_ * C_;
  __bf16* twb  = rwb + CO_ * C_;              // [o][kh][c], 147456 elems
  float*  tpre = y;                           // alias: y dead after k_gcn

  dim3 blk(256);
  k_cvt<<<131072, blk, 0, stream>>>(x, xc, xbt);
  k_w2bf<<<24, blk, 0, stream>>>(wa, wab, 3 * IC_ * C_);
  k_w2bf<<<24, blk, 0, stream>>>(wb, wbb, 3 * IC_ * C_);
  k_w2bf<<<96, blk, 0, stream>>>(wd, wdb, 3 * CO_ * C_);
  k_w2bf<<<32, blk, 0, stream>>>(cres_w, cwb, CO_ * C_);
  k_w2bf<<<32, blk, 0, stream>>>(rt_w, rwb, CO_ * C_);
  k_twcvt<<<576, blk, 0, stream>>>(tcn_w, twb);

  for (int i = 0; i < 3; ++i) {
    k_embed<<<8192, blk, 0, stream>>>(wab + i * IC_ * C_, ba + i * IC_, xc, ap);
    k_embed<<<8192, blk, 0, stream>>>(wbb + i * IC_ * C_, bb + i * IC_, xc, bp);
    k_scores<<<256, blk, 0, stream>>>(ap, bp, sc);
    k_softmax<<<2048, blk, 0, stream>>>(sc, mat_adj + i * TV_, adj_w + i * TV_, adT);
    k_xa<<<16384, blk, 0, stream>>>(xbt, adT, xa);
    k_y<<<32768, blk, 0, stream>>>(wdb + i * CO_ * C_, bd + i * CO_, xa, y, i);
  }
  k_gcn<<<32768, blk, 0, stream>>>(cwb, cres_b, gcn_bn, cres_bn, xc, y, gcn);
  k_tcn<<<32768, blk, 0, stream>>>(twb, gcn, tpre);
  k_final<<<32768, blk, 0, stream>>>(rwb, rt_b, tcn_bn, tcn_b, rt_bn, xc, tpre, out);
}
